// MultiHeadAttention_27101243638291
// MI455X (gfx1250) — matrix operand order repaired and checked by reading, not compiled, not run on hardware
//
#include <hip/hip_runtime.h>
#include <stdint.h>
#include <stddef.h>

#define SEQ     4096
#define DMODEL  512
#define NHEADS  8
#define DK      64

typedef __attribute__((ext_vector_type(16))) __bf16 bf16x16;
typedef __attribute__((ext_vector_type(8)))  float  f32x8;
typedef __attribute__((ext_vector_type(4)))  unsigned int u32x4;
typedef __attribute__((ext_vector_type(8)))  int    i32x8;
typedef __attribute__((ext_vector_type(4)))  int    i32x4;

#if defined(__gfx1250__) && __has_builtin(__builtin_amdgcn_tensor_load_to_lds)
#define HAS_TDM 1
#else
#define HAS_TDM 0
#endif

#define WMMA_BF16(a, b, c) \
  __builtin_amdgcn_wmma_f32_16x16x32_bf16(false, (a), false, (b), (short)0, (c), false, false)

// fp32 -> bf16, round-to-nearest-even
__device__ __forceinline__ unsigned short f2bf(float x) {
  unsigned u = __float_as_uint(x);
  u += 0x7FFFu + ((u >> 16) & 1u);
  return (unsigned short)(u >> 16);
}

// A-fragment (16x32 bf16, MxK) from row-major LDS tile [row][k], stride ld.
__device__ __forceinline__ bf16x16 frag_a_lds(const unsigned short* base, int row0, int k0, int ld) {
  const int lane = threadIdx.x & 31;
  const int row  = row0 + (lane & 15);
  const int kb   = k0 + ((lane & 16) ? 16 : 0);
  union { bf16x16 v; unsigned u[8]; } f;
  const unsigned short* p = base + row * ld + kb;
#pragma unroll
  for (int i = 0; i < 8; ++i) {
    const int k = 2 * i;
    f.u[i] = *(const unsigned*)(p + k);
  }
  return f.v;
}

// B-fragment (32x16 bf16, KxN) from LDS tile stored [N][K], stride ld.
__device__ __forceinline__ bf16x16 frag_b_lds(const unsigned short* base, int n0, int k0, int ld) {
  const int lane = threadIdx.x & 31;
  const int n  = n0 + (lane & 15);
  const int kb = k0 + ((lane & 16) ? 16 : 0);
  union { bf16x16 v; unsigned u[8]; } f;
  const unsigned short* p = base + n * ld + kb;
#pragma unroll
  for (int i = 0; i < 8; ++i) f.u[i] = *(const unsigned*)(p + 2 * i);
  return f.v;
}

#if HAS_TDM
// TDM 2D tile load global->LDS, bf16 elements, row pad 32DW->+4DW (LDS stride 72 shorts).
// dims/tiles/stride in elements. Issued once by the calling wave; tracked by TENSORcnt.
__device__ __forceinline__ void tdm_load_2d(unsigned lds_byte_addr, const void* gaddr,
                                            unsigned dim0, unsigned dim1,
                                            unsigned tile0, unsigned tile1,
                                            unsigned stride0) {
  const unsigned long long ga = (unsigned long long)gaddr;
  u32x4 g0;
  g0[0] = 1u;                                            // count=1 (user mode)
  g0[1] = lds_byte_addr;                                 // lds_addr [63:32]
  g0[2] = (unsigned)(ga & 0xFFFFFFFFu);                  // global_addr [95:64]
  g0[3] = (unsigned)((ga >> 32) & 0x01FFFFFFu) | (2u << 30); // addr[56:32] | type=2
  i32x8 g1;
  g1[0] = (int)((1u << 16) | (1u << 20) | (4u << 22) | (3u << 25)); // 2B, pad_en, every 32DW, +4DW
  g1[1] = (int)(dim0 << 16);                             // tensor_dim0[15:0] -> bits 63:48
  g1[2] = (int)((dim0 >> 16) | (dim1 << 16));            // dim0 hi | dim1 lo
  g1[3] = (int)((dim1 >> 16) | (tile0 << 16));           // dim1 hi | tile_dim0
  g1[4] = (int)(tile1 & 0xFFFFu);                        // tile_dim1 (tile_dim2 = 0)
  g1[5] = (int)stride0;                                  // tensor_dim0_stride[31:0]
  g1[6] = 0;
  g1[7] = 0;
  const i32x4 z4 = {0, 0, 0, 0};
  const i32x8 z8 = {0, 0, 0, 0, 0, 0, 0, 0};
  __builtin_amdgcn_tensor_load_to_lds(g0, g1, z4, z4, z8, 0);
}
#endif

// ---------------------------------------------------------------------------
// Projection GEMM: Y[8192, 512] = X(fp32) @ W(fp32) + bias.
// mode 0: fp32 row-major out_f[M,512]
// mode 1: bf16 out_bf [B, H, L, 64]
// mode 2: bf16 out_bf [B, H, 64, L]   (pre-transposed V for TDM tile loads)
// ---------------------------------------------------------------------------
__global__ __launch_bounds__(256)
void proj_kernel(const float* __restrict__ X, const float* __restrict__ W,
                 const float* __restrict__ bias,
                 unsigned short* __restrict__ out_bf, float* __restrict__ out_f,
                 int mode) {
  constexpr int LD = 72;
  __shared__ unsigned short As[128 * LD];
  __shared__ unsigned short Bs[64 * LD];

  const int tid  = threadIdx.x;
  const int w    = tid >> 5;
  const int lane = tid & 31;
  const int r0   = blockIdx.x * 128;
  const int n0   = blockIdx.y * 64;

  f32x8 acc[4];
#pragma unroll
  for (int t = 0; t < 4; ++t)
#pragma unroll
    for (int e = 0; e < 8; ++e) acc[t][e] = 0.0f;

  for (int kc = 0; kc < DMODEL; kc += 64) {
    __syncthreads();
    // X tile 128x64 fp32 -> bf16 (float4 loads, 8B LDS stores)
    for (int i = tid; i < 128 * 16; i += 256) {
      const int rr = i >> 4, c4 = (i & 15) << 2;
      const float4 xv = *(const float4*)&X[(size_t)(r0 + rr) * DMODEL + kc + c4];
      uint2 pk;
      pk.x = (unsigned)f2bf(xv.x) | ((unsigned)f2bf(xv.y) << 16);
      pk.y = (unsigned)f2bf(xv.z) | ((unsigned)f2bf(xv.w) << 16);
      *(uint2*)&As[rr * LD + c4] = pk;
    }
    // W tile 64x64 transposed: Bs[n][k]
    for (int i = tid; i < 64 * 16; i += 256) {
      const int kk = i >> 4, n4 = (i & 15) << 2;
      const float4 wv = *(const float4*)&W[(size_t)(kc + kk) * DMODEL + n0 + n4];
      Bs[(n4 + 0) * LD + kk] = f2bf(wv.x);
      Bs[(n4 + 1) * LD + kk] = f2bf(wv.y);
      Bs[(n4 + 2) * LD + kk] = f2bf(wv.z);
      Bs[(n4 + 3) * LD + kk] = f2bf(wv.w);
    }
    __syncthreads();

    const bf16x16 a0 = frag_a_lds(As, w * 16, 0, LD);
    const bf16x16 a1 = frag_a_lds(As, w * 16, 32, LD);
#pragma unroll
    for (int t = 0; t < 4; ++t) {
      const bf16x16 b0 = frag_b_lds(Bs, t * 16, 0, LD);
      const bf16x16 b1 = frag_b_lds(Bs, t * 16, 32, LD);
      acc[t] = WMMA_BF16(a0, b0, acc[t]);
      acc[t] = WMMA_BF16(a1, b1, acc[t]);
    }
  }

  const int hi = (lane & 16) ? 8 : 0;
  const int cl = lane & 15;
  const int h  = blockIdx.y; // head index for modes 1/2 (N-tile == dk == 64)
#pragma unroll
  for (int t = 0; t < 4; ++t) {
    const int n = n0 + t * 16 + cl;
    const float bv = bias[n];
#pragma unroll
    for (int v = 0; v < 8; ++v) {
      const int m = r0 + w * 16 + hi + v;
      const float val = acc[t][v] + bv;
      const int b = m >> 12;        // / SEQ
      const int l = m & (SEQ - 1);  // % SEQ
      const int d = t * 16 + cl;
      if (mode == 1) {
        out_bf[((size_t)(b * NHEADS + h) * SEQ + l) * DK + d] = f2bf(val);
      } else if (mode == 2) {
        out_bf[((size_t)(b * NHEADS + h) * DK + d) * SEQ + l] = f2bf(val);
      } else {
        out_f[(size_t)m * DMODEL + n] = val;
      }
    }
  }
}

// ---------------------------------------------------------------------------
// Flash attention: per (b,h), 128-query block (8 waves x 16 rows), KV steps of 64.
// Qh/Kh: bf16 [B*H, L, 64]; Vt: bf16 [B*H, 64, L]; ctx: fp32 [B, L, H, 64].
// Dynamic LDS layout (base offset 0, stride-72 tiles):
//   Ps:   [0, 18432)          128x72 bf16  (Q staging, then per-wave P scratch)
//   K[2]: [18432, 36864)      2 x 64x72 bf16
//   V[2]: [36864, 55296)      2 x 64x72 bf16
// ---------------------------------------------------------------------------
#define ATTN_LDS_BYTES 55296

__global__ __launch_bounds__(256)
void attn_kernel(const unsigned short* __restrict__ Qh,
                 const unsigned short* __restrict__ Kh,
                 const unsigned short* __restrict__ Vt,
                 float* __restrict__ ctx) {
  constexpr int LD = 72;
  constexpr int TILE = 64 * LD;                // elements per K/V tile
  extern __shared__ unsigned short smem[];
  unsigned short* Ps  = smem;                  // 128*LD
  unsigned short* Ksb = smem + 128 * LD;
  unsigned short* Vsb = smem + 128 * LD + 2 * TILE;
  const unsigned off_K0 = 128 * LD * 2;        // bytes
  const unsigned off_V0 = off_K0 + 2 * TILE * 2;

  const int tid  = threadIdx.x;
  const int w    = tid >> 5;
  const int lane = tid & 31;
  const int bh   = blockIdx.y;
  const int bb   = bh >> 3;
  const int h    = bh & 7;
  const int qr0  = blockIdx.x * 128;

  const unsigned short* Qg = Qh + (size_t)bh * SEQ * DK;
  const unsigned short* Kg = Kh + (size_t)bh * SEQ * DK;
  const unsigned short* Vg = Vt + (size_t)bh * DK * SEQ;

#if HAS_TDM
  if (w == 0) {
    // Q first, then K0/V0: wait(<=2) below means "Q has landed".
    tdm_load_2d(0u, Qg + (size_t)qr0 * DK, DK, SEQ, 64, 128, DK);
    tdm_load_2d(off_K0, Kg, DK, SEQ, 64, 64, DK);
    tdm_load_2d(off_V0, Vg, SEQ, DK, 64, 64, SEQ);
    __builtin_amdgcn_s_wait_tensorcnt(2);
  }
  __syncthreads(); // Q tile visible to all waves
#else
  for (int i = tid; i < 128 * 8; i += 256) {
    const int rr = i >> 3, c8 = (i & 7) << 3;
    *(uint4*)&Ps[rr * LD + c8] = *(const uint4*)&Qg[(size_t)(qr0 + rr) * DK + c8];
  }
  __syncthreads();
#endif

  const bf16x16 qa0 = frag_a_lds(Ps, w * 16, 0, LD);
  const bf16x16 qa1 = frag_a_lds(Ps, w * 16, 32, LD);

  f32x8 acc[4];
  float m_i[8], l_i[8];
#pragma unroll
  for (int v = 0; v < 8; ++v) { m_i[v] = -3.0e38f; l_i[v] = 0.0f; }
#pragma unroll
  for (int t = 0; t < 4; ++t)
#pragma unroll
    for (int e = 0; e < 8; ++e) acc[t][e] = 0.0f;

  const int hi = (lane & 16) ? 8 : 0;
  const int cl = lane & 15;
  const float scale = 0.125f; // 1/sqrt(64)

  for (int j0 = 0; j0 < SEQ; j0 += 64) {
    const int cur = (j0 >> 6) & 1;
    const unsigned short* Ksc = Ksb + cur * TILE;
    const unsigned short* Vsc = Vsb + cur * TILE;

#if HAS_TDM
    if (w == 0) __builtin_amdgcn_s_wait_tensorcnt(0); // tile `cur` landed
    __syncthreads(); // all waves: tile cur ready, tile cur^1 free
    if (w == 0 && j0 + 64 < SEQ) {
      const unsigned nx = (unsigned)(cur ^ 1);
      tdm_load_2d(off_K0 + nx * TILE * 2, Kg + (size_t)(j0 + 64) * DK, DK, SEQ, 64, 64, DK);
      tdm_load_2d(off_V0 + nx * TILE * 2, Vg + (j0 + 64), SEQ, DK, 64, 64, SEQ);
    }
#else
    __syncthreads(); // previous iteration done reading tiles
    unsigned short* Kw = Ksb + cur * TILE;
    unsigned short* Vw = Vsb + cur * TILE;
    for (int i = tid; i < 64 * 8; i += 256) {
      const int rr = i >> 3, c8 = (i & 7) << 3;
      *(uint4*)&Kw[rr * LD + c8] = *(const uint4*)&Kg[(size_t)(j0 + rr) * DK + c8];
      *(uint4*)&Vw[rr * LD + c8] = *(const uint4*)&Vg[(size_t)rr * SEQ + j0 + c8];
    }
    __syncthreads();
#endif

    // S = Q K^T
    f32x8 s[4];
#pragma unroll
    for (int t = 0; t < 4; ++t) {
#pragma unroll
      for (int e = 0; e < 8; ++e) s[t][e] = 0.0f;
      const bf16x16 b0 = frag_b_lds(Ksc, t * 16, 0, LD);
      const bf16x16 b1 = frag_b_lds(Ksc, t * 16, 32, LD);
      s[t] = WMMA_BF16(qa0, b0, s[t]);
      s[t] = WMMA_BF16(qa1, b1, s[t]);
    }

    // Online softmax per row v (row lives in one 16-lane half).
#pragma unroll
    for (int v = 0; v < 8; ++v) {
      float mx = -3.0e38f;
#pragma unroll
      for (int t = 0; t < 4; ++t) {
        s[t][v] *= scale;
        mx = fmaxf(mx, s[t][v]);
      }
#pragma unroll
      for (int msk = 1; msk < 16; msk <<= 1) mx = fmaxf(mx, __shfl_xor(mx, msk));
      const float mn = fmaxf(m_i[v], mx);
      const float sc = __expf(m_i[v] - mn);
      float rs = 0.0f;
#pragma unroll
      for (int t = 0; t < 4; ++t) {
        const float p = __expf(s[t][v] - mn);
        s[t][v] = p;
        rs += p;
      }
#pragma unroll
      for (int msk = 1; msk < 16; msk <<= 1) rs += __shfl_xor(rs, msk);
      l_i[v] = l_i[v] * sc + rs;
      m_i[v] = mn;
#pragma unroll
      for (int t = 0; t < 4; ++t) acc[t][v] *= sc;
    }

    // P -> wave-private LDS rows (bf16); HW DS ops are in-order per wave,
    // the asm barrier stops the compiler reordering the re-gather above the stores.
#pragma unroll
    for (int t = 0; t < 4; ++t)
#pragma unroll
      for (int v = 0; v < 8; ++v)
        Ps[(w * 16 + hi + v) * LD + t * 16 + cl] = f2bf(s[t][v]);
    asm volatile("" ::: "memory");

    // ctx += P @ V
    const bf16x16 pa0 = frag_a_lds(Ps, w * 16, 0, LD);
    const bf16x16 pa1 = frag_a_lds(Ps, w * 16, 32, LD);
#pragma unroll
    for (int t = 0; t < 4; ++t) {
      const bf16x16 vb0 = frag_b_lds(Vsc, t * 16, 0, LD);
      const bf16x16 vb1 = frag_b_lds(Vsc, t * 16, 32, LD);
      acc[t] = WMMA_BF16(pa0, vb0, acc[t]);
      acc[t] = WMMA_BF16(pa1, vb1, acc[t]);
    }
  }

  // Normalize and write ctx fp32 as [B, L, H, 64]
#pragma unroll
  for (int v = 0; v < 8; ++v) {
    const float inv = 1.0f / l_i[v];
    const int qrow = qr0 + w * 16 + hi + v;
    const size_t base = ((size_t)(bb * SEQ + qrow) * NHEADS + h) * DK;
#pragma unroll
    for (int t = 0; t < 4; ++t)
      ctx[base + t * 16 + cl] = acc[t][v] * inv;
  }
}

// ---------------------------------------------------------------------------
extern "C" void kernel_launch(void* const* d_in, const int* in_sizes, int n_in,
                              void* d_out, int out_size, void* d_ws, size_t ws_size,
                              hipStream_t stream) {
  const float* q  = (const float*)d_in[0];
  const float* k  = (const float*)d_in[1];
  const float* v  = (const float*)d_in[2];
  const float* Wq = (const float*)d_in[3];
  const float* bq = (const float*)d_in[4];
  const float* Wk = (const float*)d_in[5];
  const float* bk = (const float*)d_in[6];
  const float* Wv = (const float*)d_in[7];
  const float* bv = (const float*)d_in[8];
  const float* Wo = (const float*)d_in[9];
  const float* bo = (const float*)d_in[10];

  const size_t elems = (size_t)2 * NHEADS * SEQ * DK; // 4M elems per head-tensor
  unsigned short* Qh = (unsigned short*)d_ws;
  unsigned short* Kh = Qh + elems;
  unsigned short* Vt = Kh + elems;
  float* ctx = (float*)(Vt + elems);

  dim3 blk(256, 1, 1);
  dim3 gproj(8192 / 128, DMODEL / 64, 1); // (64, 8)
  dim3 gattn(SEQ / 128, 2 * NHEADS, 1);   // (32, 16)

  proj_kernel<<<gproj, blk, 0, stream>>>(q, Wq, bq, Qh, nullptr, 1);
  proj_kernel<<<gproj, blk, 0, stream>>>(k, Wk, bk, Kh, nullptr, 1);
  proj_kernel<<<gproj, blk, 0, stream>>>(v, Wv, bv, Vt, nullptr, 2);
  attn_kernel<<<gattn, blk, ATTN_LDS_BYTES, stream>>>(Qh, Kh, Vt, ctx);
  proj_kernel<<<gproj, blk, 0, stream>>>(ctx, Wo, bo, nullptr, (float*)d_out, 0);
}
